// SelfAttention_66829691126494
// MI455X (gfx1250) — matrix operand order repaired and checked by reading, not compiled, not run on hardware
//
#include <hip/hip_runtime.h>
#include <hip/hip_bf16.h>

typedef __attribute__((ext_vector_type(16))) _Float16 v16h;
typedef __attribute__((ext_vector_type(8)))  _Float16 v8h;
typedef __attribute__((ext_vector_type(8)))  float    v8f;
typedef __attribute__((ext_vector_type(4)))  int      v4i;

#define B_  4
#define T_  2048
#define C_  768
#define H_  16
#define D_  48
#define DP_ 64
#define M_  (B_*T_)

#if defined(__HIP_DEVICE_COMPILE__) && __has_builtin(__builtin_amdgcn_global_load_async_to_lds_b128)
#define HAVE_ASYNC_LDS 1
typedef __attribute__((address_space(1))) v4i g_v4i;   // global (AS1) int4
typedef __attribute__((address_space(3))) v4i l_v4i;   // LDS    (AS3) int4
#endif

static __device__ __forceinline__ v16h pack16(v8h lo, v8h hi) {
  v16h r;
#pragma unroll
  for (int i = 0; i < 8; ++i) { r[i] = lo[i]; r[i + 8] = hi[i]; }
  return r;
}

// ---------------------------------------------------------------- f32 -> f16
__global__ void cvt_f32_f16(const float* __restrict__ src, _Float16* __restrict__ dst, int n) {
  int i = blockIdx.x * blockDim.x + threadIdx.x;
  if (i < n) dst[i] = (_Float16)src[i];
}

// ------------------------------------------------- C[M,N] = A[M,K] * W[N,K]^T + bias
// block = 256 threads (8 waves, 4x2), tile 128(M) x 128(N), wave tile 32x64, K step 32.
template <typename OutT>
__global__ __launch_bounds__(256)
void gemm_xwt(const _Float16* __restrict__ A, const _Float16* __restrict__ W,
              const float* __restrict__ bias, OutT* __restrict__ Cmat,
              int M, int N, int K) {
  __shared__ __align__(32) _Float16 As[128 * 32];
  __shared__ __align__(32) _Float16 Bs[128 * 32];
  const int tid  = threadIdx.x;
  const int lane = tid & 31, wv = tid >> 5;
  const int wm = wv >> 1, wn = wv & 1;             // 4 x 2 wave grid
  const int m0 = blockIdx.y * 128, n0 = blockIdx.x * 128;
  const int srow = tid >> 1, sc16 = (tid & 1) * 16;   // stage: 32B per thread per tile
  v8f acc[2][4] = {};

  for (int k0 = 0; k0 < K; k0 += 32) {
#if defined(HAVE_ASYNC_LDS)
    {  // ASYNCcnt path: DMA 16B x2 straight into LDS, no VGPR staging
      g_v4i* ga = (g_v4i*)&A[(size_t)(m0 + srow) * K + k0 + sc16];
      g_v4i* gw = (g_v4i*)&W[(size_t)(n0 + srow) * K + k0 + sc16];
      l_v4i* la = (l_v4i*)&As[srow * 32 + sc16];
      l_v4i* lw = (l_v4i*)&Bs[srow * 32 + sc16];
      __builtin_amdgcn_global_load_async_to_lds_b128(ga, la, 0, 0);
      __builtin_amdgcn_global_load_async_to_lds_b128(ga, la, 16, 0);
      __builtin_amdgcn_global_load_async_to_lds_b128(gw, lw, 0, 0);
      __builtin_amdgcn_global_load_async_to_lds_b128(gw, lw, 16, 0);
    }
#else
    *(v16h*)&As[srow * 32 + sc16] =
        *(const v16h*)&A[(size_t)(m0 + srow) * K + k0 + sc16];
    *(v16h*)&Bs[srow * 32 + sc16] =
        *(const v16h*)&W[(size_t)(n0 + srow) * K + k0 + sc16];
#endif
    if (k0 + 32 < K) {   // pull next K-tile toward L2 (global_prefetch_b8)
      __builtin_prefetch(&A[(size_t)(m0 + srow) * K + k0 + 32 + sc16], 0, 0);
      __builtin_prefetch(&W[(size_t)(n0 + srow) * K + k0 + 32 + sc16], 0, 0);
    }
#if defined(HAVE_ASYNC_LDS)
    asm volatile("s_wait_asynccnt 0" ::: "memory");
#endif
    __syncthreads();

    v16h af[2], bf[4];
#pragma unroll
    for (int mi = 0; mi < 2; ++mi) {      // A-frag: row=lane%16, K=16*(lane/16)+e
      int r = wm * 32 + mi * 16 + (lane & 15);
      int cb = (lane >> 4) * 16;
      af[mi] = *(const v16h*)&As[r * 32 + cb];
    }
#pragma unroll
    for (int ni = 0; ni < 4; ++ni) {      // B-frag: col=lane%16, K=16*(lane/16)+e
      int r = wn * 64 + ni * 16 + (lane & 15);
      bf[ni] = *(const v16h*)&Bs[r * 32 + (lane >> 4) * 16];
    }
#pragma unroll
    for (int mi = 0; mi < 2; ++mi)
#pragma unroll
      for (int ni = 0; ni < 4; ++ni)
        acc[mi][ni] = __builtin_amdgcn_wmma_f32_16x16x32_f16(
            false, af[mi], false, bf[ni], (short)0, acc[mi][ni], false, false);
    __syncthreads();
  }

#pragma unroll
  for (int mi = 0; mi < 2; ++mi)
#pragma unroll
    for (int ni = 0; ni < 4; ++ni) {
      int ng = n0 + wn * 64 + ni * 16 + (lane & 15);
      float bb = bias[ng];
#pragma unroll
      for (int r = 0; r < 8; ++r) {       // C-frag: row = r + 8*(lane/16), col = lane%16
        int mg = m0 + wm * 32 + mi * 16 + r + (lane >> 4) * 8;
        Cmat[(size_t)mg * N + ng] = (OutT)(acc[mi][ni][r] + bb);
      }
    }
}

// ------------------------------------- RoPE + head split + pad d 48->64, f16 out
__global__ void rope_pack(const _Float16* __restrict__ Qr, const _Float16* __restrict__ Kr,
                          _Float16* __restrict__ Qp, _Float16* __restrict__ Kp) {
  int idx = blockIdx.x * blockDim.x + threadIdx.x;     // [bh][t][dp]
  int dp = idx & (DP_ - 1);
  int t  = (idx >> 6) & (T_ - 1);
  int bh = idx >> 17;
  if (bh >= B_ * H_) return;
  size_t po = (size_t)idx;
  if (dp >= D_) { Qp[po] = (_Float16)0.f; Kp[po] = (_Float16)0.f; return; }
  int h = bh & (H_ - 1), b = bh >> 4;
  size_t ro = ((size_t)(b * T_ + t)) * C_ + h * D_ + dp;
  int fi = dp % (D_ / 2);
  float invf = __expf(-(2.0f * (float)fi / (float)D_) * 9.210340371976184f); // ln(1e4)
  float ang = (float)t * invf;
  float s, c;
  __sincosf(ang, &s, &c);
  size_t ro2 = (dp < D_ / 2) ? ro + D_ / 2 : ro - D_ / 2;
  float sgn = (dp < D_ / 2) ? -1.f : 1.f;
  float q = (float)Qr[ro], k = (float)Kr[ro];
  float qr = sgn * (float)Qr[ro2], kr = sgn * (float)Kr[ro2];
  Qp[po] = (_Float16)(q * c + qr * s);
  Kp[po] = (_Float16)(k * c + kr * s);
}

// ------------------------------------------------- V -> Vt[b,h,d,t] (f16)
__global__ void v_pack(const _Float16* __restrict__ Vr, _Float16* __restrict__ Vt) {
  int idx = blockIdx.x * blockDim.x + threadIdx.x;     // [bh][d][t]
  if (idx >= B_ * H_ * D_ * T_) return;
  int t  = idx & (T_ - 1);
  int dd = idx >> 11;
  int d  = dd % D_;
  int bh = dd / D_;
  int h = bh & (H_ - 1), b = bh >> 4;
  Vt[idx] = Vr[((size_t)(b * T_ + t)) * C_ + h * D_ + d];
}

// ------------------------------------------------- flash attention, 4 waves/block
__global__ __launch_bounds__(128)
void attn(const _Float16* __restrict__ Qp, const _Float16* __restrict__ Kp,
          const _Float16* __restrict__ Vt, const float* __restrict__ amask,
          _Float16* __restrict__ ctx) {
  __shared__ __align__(32) _Float16 Pl[4][16 * 32];    // per-wave P staging
  const int lane = threadIdx.x & 31, wv = threadIdx.x >> 5;
  const int bh = blockIdx.y, b = bh >> 4, h = bh & (H_ - 1);
  const int q0 = blockIdx.x * 64 + wv * 16;
  const float scale = 0.14433756729740643f;            // 1/sqrt(48)

  v16h aq[2];
#pragma unroll
  for (int kc = 0; kc < 2; ++kc) {                     // Q A-frags, 2 d-chunks
    size_t o = ((size_t)bh * T_ + q0 + (lane & 15)) * DP_ + kc * 32 + (lane >> 4) * 16;
    aq[kc] = *(const v16h*)&Qp[o];
  }

  v8f acc[3] = {};
  float mrow[8], lrow[8];
#pragma unroll
  for (int r = 0; r < 8; ++r) { mrow[r] = -INFINITY; lrow[r] = 0.f; }

  const int nkt = (q0 + 47) >> 5;                      // k0 <= q0 always
  for (int kt = 0; kt < nkt; ++kt) {
    const int k0 = kt * 32;
    v8f s[2] = {};
    int colg[2];
    float mb[2];
#pragma unroll
    for (int ni = 0; ni < 2; ++ni) {
      colg[ni] = k0 + ni * 16 + (lane & 15);
      int kc_ = colg[ni] < T_ ? colg[ni] : T_ - 1;
      mb[ni] = (1.0f - amask[b * T_ + kc_]) * -10000.0f;
#pragma unroll
      for (int kc = 0; kc < 2; ++kc) {                 // K B-frag: col=kpos, contiguous d
        v16h bk = *(const v16h*)&Kp[((size_t)bh * T_ + kc_) * DP_ + kc * 32 + (lane >> 4) * 16];
        s[ni] = __builtin_amdgcn_wmma_f32_16x16x32_f16(
            false, aq[kc], false, bk, (short)0, s[ni], false, false);
      }
    }
    // online softmax over the 16x32 score block
    float p0[8], p1[8];
#pragma unroll
    for (int r = 0; r < 8; ++r) {
      int rowg = q0 + r + (lane >> 4) * 8;
      float v0 = s[0][r] * scale + mb[0];
      float v1 = s[1][r] * scale + mb[1];
      if (colg[0] > rowg) v0 = -INFINITY;
      if (colg[1] > rowg) v1 = -INFINITY;
      float mx = fmaxf(v0, v1);
#pragma unroll
      for (int off = 8; off >= 1; off >>= 1) mx = fmaxf(mx, __shfl_xor(mx, off));
      float mnew = fmaxf(mrow[r], mx);
      float corr = __expf(mrow[r] - mnew);
      float e0 = __expf(v0 - mnew), e1 = __expf(v1 - mnew);
      float sum = e0 + e1;
#pragma unroll
      for (int off = 8; off >= 1; off >>= 1) sum += __shfl_xor(sum, off);
      lrow[r] = lrow[r] * corr + sum;
      mrow[r] = mnew;
      p0[r] = e0; p1[r] = e1;
#pragma unroll
      for (int j = 0; j < 3; ++j) acc[j][r] *= corr;
    }
    // C-frag layout -> row-major LDS -> A-frag layout (wave-private, in-order DS)
#pragma unroll
    for (int r = 0; r < 8; ++r) {
      int row = r + (lane >> 4) * 8;
      Pl[wv][row * 32 + (lane & 15)]      = (_Float16)p0[r];
      Pl[wv][row * 32 + 16 + (lane & 15)] = (_Float16)p1[r];
    }
    __builtin_amdgcn_wave_barrier();
    asm volatile("s_wait_dscnt 0" ::: "memory");
    {
      int row = lane & 15, cb = (lane >> 4) * 16;
      v16h pf = *(const v16h*)&Pl[wv][row * 32 + cb];
#pragma unroll
      for (int j = 0; j < 3; ++j) {                    // V B-frag from Vt[b,h,d,t]
        int d = j * 16 + (lane & 15);
        v16h vf = *(const v16h*)&Vt[((size_t)bh * D_ + d) * T_ + k0 + (lane >> 4) * 16];
        acc[j] = __builtin_amdgcn_wmma_f32_16x16x32_f16(
            false, pf, false, vf, (short)0, acc[j], false, false);
      }
    }
    __builtin_amdgcn_wave_barrier();
  }

#pragma unroll
  for (int j = 0; j < 3; ++j) {
    int d = j * 16 + (lane & 15);
#pragma unroll
    for (int r = 0; r < 8; ++r) {
      int rowg = q0 + r + (lane >> 4) * 8;
      ctx[((size_t)(b * T_ + rowg)) * C_ + h * D_ + d] = (_Float16)(acc[j][r] / lrow[r]);
    }
  }
}

// ----------------------------------------------------------------------------
extern "C" void kernel_launch(void* const* d_in, const int* in_sizes, int n_in,
                              void* d_out, int out_size, void* d_ws, size_t ws_size,
                              hipStream_t stream) {
  (void)in_sizes; (void)n_in; (void)out_size; (void)ws_size;
  const float* x  = (const float*)d_in[0];
  const float* am = (const float*)d_in[1];
  const float* Wq = (const float*)d_in[2];
  const float* bq = (const float*)d_in[3];
  const float* Wk = (const float*)d_in[4];
  const float* bk = (const float*)d_in[5];
  const float* Wv = (const float*)d_in[6];
  const float* bv = (const float*)d_in[7];
  const float* Wo = (const float*)d_in[8];
  const float* bo = (const float*)d_in[9];
  float* out = (float*)d_out;

  char* ws = (char*)d_ws;
  size_t off = 0;
  auto alloc = [&](size_t bytes) -> char* {
    char* p = ws + off;
    off = (off + bytes + 255) & ~(size_t)255;
    return p;
  };
  _Float16* Xh   = (_Float16*)alloc((size_t)M_ * C_ * 2);
  _Float16* Wqh  = (_Float16*)alloc((size_t)C_ * C_ * 2);
  _Float16* Wkh  = (_Float16*)alloc((size_t)C_ * C_ * 2);
  _Float16* Wvh  = (_Float16*)alloc((size_t)C_ * C_ * 2);
  _Float16* Woh  = (_Float16*)alloc((size_t)C_ * C_ * 2);
  _Float16* Qr   = (_Float16*)alloc((size_t)M_ * C_ * 2);
  _Float16* Kr   = (_Float16*)alloc((size_t)M_ * C_ * 2);
  _Float16* Vr   = (_Float16*)alloc((size_t)M_ * C_ * 2);
  _Float16* Qp   = (_Float16*)alloc((size_t)B_ * H_ * T_ * DP_ * 2);
  _Float16* Kp   = (_Float16*)alloc((size_t)B_ * H_ * T_ * DP_ * 2);
  _Float16* Vt   = (_Float16*)alloc((size_t)B_ * H_ * D_ * T_ * 2 + 256); // +OOB pad
  _Float16* Ctxh = Qr;   // Qr dead after rope_pack; reuse for ctx

  const int nX = M_ * C_, nW = C_ * C_;
  cvt_f32_f16<<<(nX + 255) / 256, 256, 0, stream>>>(x,  Xh,  nX);
  cvt_f32_f16<<<(nW + 255) / 256, 256, 0, stream>>>(Wq, Wqh, nW);
  cvt_f32_f16<<<(nW + 255) / 256, 256, 0, stream>>>(Wk, Wkh, nW);
  cvt_f32_f16<<<(nW + 255) / 256, 256, 0, stream>>>(Wv, Wvh, nW);
  cvt_f32_f16<<<(nW + 255) / 256, 256, 0, stream>>>(Wo, Woh, nW);

  dim3 gg(C_ / 128, M_ / 128);
  gemm_xwt<_Float16><<<gg, 256, 0, stream>>>(Xh, Wqh, bq, Qr, M_, C_, C_);
  gemm_xwt<_Float16><<<gg, 256, 0, stream>>>(Xh, Wkh, bk, Kr, M_, C_, C_);
  gemm_xwt<_Float16><<<gg, 256, 0, stream>>>(Xh, Wvh, bv, Vr, M_, C_, C_);

  const int nR = B_ * H_ * T_ * DP_;
  rope_pack<<<nR / 256, 256, 0, stream>>>(Qr, Kr, Qp, Kp);
  const int nV = B_ * H_ * D_ * T_;
  v_pack<<<(nV + 255) / 256, 256, 0, stream>>>(Vr, Vt);

  dim3 ga(T_ / 64, B_ * H_);
  attn<<<ga, 128, 0, stream>>>(Qp, Kp, Vt, am, Ctxh);

  gemm_xwt<float><<<gg, 256, 0, stream>>>(Ctxh, Woh, bo, out, M_, C_, C_);
}
